// HydraBlock_10771777978688
// MI455X (gfx1250) — matrix operand order repaired and checked by reading, not compiled, not run on hardware
//
#include <hip/hip_runtime.h>

// ---------------- problem dimensions ----------------
#define BATCHN   2
#define SEQLEN   4096
#define DMODEL   768
#define DSTATE   64
#define DCONV    7
#define HEADDIM  64
#define NHEADS   24
#define DINNER   1536
#define DINPROJ  3248          // 2*1536 + 2*64 + 2*24
#define CONVDIM  1664          // 1536 + 128
#define CHUNK    256
#define NCHUNK   16            // SEQLEN / CHUNK
#define NB2      4             // 2 * BATCHN (fwd + flipped)
#define ROWS     8192          // BATCHN * SEQLEN

typedef __attribute__((ext_vector_type(16))) __bf16 bf16x16;
typedef __attribute__((ext_vector_type(8)))  float  f32x8;

// =====================================================================
// LayerNorm over D_MODEL (one block per row)
// =====================================================================
__global__ __launch_bounds__(256) void k_layernorm(
    const float* __restrict__ x, const float* __restrict__ w,
    const float* __restrict__ b, float* __restrict__ u)
{
  int row = blockIdx.x;
  const float* xr = x + (size_t)row * DMODEL;
  float s = 0.f, s2 = 0.f;
  for (int i = threadIdx.x; i < DMODEL; i += 256) { float v = xr[i]; s += v; s2 += v * v; }
  __shared__ float shs[8], shs2[8];
  int lane = threadIdx.x & 31, wid = threadIdx.x >> 5;
  for (int o = 16; o > 0; o >>= 1) { s += __shfl_down(s, o, 32); s2 += __shfl_down(s2, o, 32); }
  if (lane == 0) { shs[wid] = s; shs2[wid] = s2; }
  __syncthreads();
  if (threadIdx.x == 0) {
    float t = 0.f, t2 = 0.f;
    for (int i = 0; i < 8; ++i) { t += shs[i]; t2 += shs2[i]; }
    shs[0] = t; shs2[0] = t2;
  }
  __syncthreads();
  float mu  = shs[0] * (1.f / DMODEL);
  float var = shs2[0] * (1.f / DMODEL) - mu * mu;
  float inv = rsqrtf(var + 1e-5f);
  float* ur = u + (size_t)row * DMODEL;
  for (int i = threadIdx.x; i < DMODEL; i += 256) ur[i] = (xr[i] - mu) * inv * w[i] + b[i];
}

// =====================================================================
// bf16 WMMA GEMM:  C[M,N] = A[M,K] * W[N,K]^T (+ R)
// block = 256 threads = 8 waves (4x2), BMxBN = 128x64, wave does 32x32
// =====================================================================
#define GBM 128
#define GBN 64
#define GBK 32
__global__ __launch_bounds__(256) void k_gemm_bf16(
    const float* __restrict__ A, const float* __restrict__ W,
    const float* __restrict__ R, float* __restrict__ C,
    int M, int N, int K)
{
  __shared__ __bf16 As[GBM][GBK + 2];
  __shared__ __bf16 Bs[GBN][GBK + 2];
  int bm = blockIdx.y * GBM, bn = blockIdx.x * GBN;
  int tid = threadIdx.x, lane = tid & 31, wid = tid >> 5;
  int wm = wid >> 1, wn = wid & 1;
  int hf = lane >> 4, l15 = lane & 15;
  f32x8 acc[2][2] = {};
  for (int k0 = 0; k0 < K; k0 += GBK) {
    for (int i = tid; i < GBM * GBK; i += 256) {
      int r = i >> 5, cc = i & 31;
      As[r][cc] = (__bf16)A[(size_t)(bm + r) * K + k0 + cc];
    }
    for (int i = tid; i < GBN * GBK; i += 256) {
      int r = i >> 5, cc = i & 31;
      int n = bn + r;
      Bs[r][cc] = (__bf16)((n < N) ? W[(size_t)n * K + k0 + cc] : 0.f);
    }
    if (k0 + GBK < K)  // warm L2 for next A tile (global_prefetch_b8)
      __builtin_prefetch(&A[(size_t)(bm + (tid >> 1)) * K + k0 + GBK], 0, 1);
    __syncthreads();
    bf16x16 afr[2], bfr[2];
#pragma unroll
    for (int mt = 0; mt < 2; ++mt) {
      int row = wm * 32 + mt * 16 + l15;
#pragma unroll
      for (int vv = 0; vv < 8; ++vv) {
        int kk = (vv < 4) ? (2 * vv + 8 * hf) : (16 + 2 * (vv - 4) + 8 * hf);
        afr[mt][2 * vv]     = As[row][kk];
        afr[mt][2 * vv + 1] = As[row][kk + 1];
      }
    }
#pragma unroll
    for (int nt = 0; nt < 2; ++nt) {
      int col = wn * 32 + nt * 16 + l15;
#pragma unroll
      for (int vv = 0; vv < 8; ++vv) {
        int kk = (vv < 4) ? (2 * vv + 8 * hf) : (16 + 2 * (vv - 4) + 8 * hf);
        bfr[nt][2 * vv]     = Bs[col][kk];
        bfr[nt][2 * vv + 1] = Bs[col][kk + 1];
      }
    }
#pragma unroll
    for (int mt = 0; mt < 2; ++mt)
#pragma unroll
      for (int nt = 0; nt < 2; ++nt)
        acc[mt][nt] = __builtin_amdgcn_wmma_f32_16x16x32_bf16(
            false, afr[mt], false, bfr[nt], (short)0, acc[mt][nt], false, false);
    __syncthreads();
  }
#pragma unroll
  for (int mt = 0; mt < 2; ++mt)
#pragma unroll
    for (int nt = 0; nt < 2; ++nt) {
      int n = bn + wn * 32 + nt * 16 + l15;
      if (n >= N) continue;
#pragma unroll
      for (int r = 0; r < 8; ++r) {
        int m = bm + wm * 32 + mt * 16 + r + 8 * hf;
        float v = acc[mt][nt][r];
        if (R) v += R[(size_t)m * N + n];
        C[(size_t)m * N + n] = v;
      }
    }
}

// =====================================================================
// Depthwise causal conv (width 7) + SiLU over xBC slice of zxbcdt
// =====================================================================
__global__ __launch_bounds__(256) void k_conv(
    const float* __restrict__ zx, const float* __restrict__ cw,
    const float* __restrict__ cb, float* __restrict__ out)
{
  int idx = blockIdx.x * 256 + threadIdx.x;
  if (idx >= BATCHN * SEQLEN * CONVDIM) return;
  int ch = idx % CONVDIM;
  int l  = (idx / CONVDIM) % SEQLEN;
  int b  = idx / (CONVDIM * SEQLEN);
  float acc = cb[ch];
#pragma unroll
  for (int t = 0; t < DCONV; ++t) {
    int ls = l - (DCONV - 1) + t;
    if (ls >= 0)
      acc += cw[ch * DCONV + t] * zx[((size_t)b * SEQLEN + ls) * DINPROJ + DINNER + ch];
  }
  out[idx] = acc / (1.f + __expf(-acc));   // silu
}

// =====================================================================
// dt = softplus(dt2 (with fwd/flip layout) + bias)   [NB2, L, NHEADS]
// =====================================================================
__global__ __launch_bounds__(256) void k_dt(
    const float* __restrict__ zx, const float* __restrict__ dt_bias,
    float* __restrict__ dt)
{
  int idx = blockIdx.x * 256 + threadIdx.x;
  if (idx >= NB2 * SEQLEN * NHEADS) return;
  int h  = idx % NHEADS;
  int l  = (idx / NHEADS) % SEQLEN;
  int b2 = idx / (NHEADS * SEQLEN);
  int sb  = (b2 < 2) ? b2 : (b2 - 2);
  int sl  = (b2 < 2) ? l : (SEQLEN - 1 - l);
  int col = DINPROJ - 2 * NHEADS + ((b2 < 2) ? h : (NHEADS + h));
  float v = zx[((size_t)sb * SEQLEN + sl) * DINPROJ + col] + dt_bias[h];
  dt[idx] = (v > 20.f) ? v : log1pf(__expf(v));
}

// =====================================================================
// Per-chunk inclusive cumsum of dt*A, one wave per (b2,c,h)
// =====================================================================
__global__ __launch_bounds__(256) void k_scan(
    const float* __restrict__ dt, const float* __restrict__ A_log,
    float* __restrict__ Acum)
{
  int gw = blockIdx.x * 8 + (threadIdx.x >> 5);
  int lane = threadIdx.x & 31;
  int h  = gw % NHEADS;
  int c  = (gw / NHEADS) % NCHUNK;
  int b2 = gw / (NHEADS * NCHUNK);
  float Ah = -__expf(A_log[h]);
  float loc[8];
  float run = 0.f;
  int lbase = c * CHUNK + lane * 8;
  for (int i = 0; i < 8; ++i) {
    float a = dt[((size_t)b2 * SEQLEN + lbase + i) * NHEADS + h] * Ah;
    run += a; loc[i] = run;
  }
  float tot = run, v = tot;
  for (int off = 1; off < 32; off <<= 1) {
    float t = __shfl_up(v, off, 32);
    if (lane >= off) v += t;
  }
  float excl = v - tot;
  float* o = Acum + (size_t)gw * CHUNK + lane * 8;
  for (int i = 0; i < 8; ++i) o[i] = excl + loc[i];
}

// =====================================================================
// Pack X = xs(±flip)*dt into bf16 [b2][h][c][l][p]; B/C states into bf16
// =====================================================================
__global__ __launch_bounds__(256) void k_pack(
    const float* __restrict__ conv, const float* __restrict__ dt,
    __bf16* __restrict__ Xbf, __bf16* __restrict__ Bm2, __bf16* __restrict__ Cm2)
{
  int b2 = blockIdx.x / SEQLEN, l = blockIdx.x % SEQLEN;
  int sb = (b2 < 2) ? b2 : (b2 - 2);
  int sl = (b2 < 2) ? l : (SEQLEN - 1 - l);
  const float* row = conv + ((size_t)sb * SEQLEN + sl) * CONVDIM;
  int c = l / CHUNK, lc = l % CHUNK;
  for (int i = threadIdx.x; i < DINNER; i += 256) {
    int h = i >> 6, p = i & 63;
    float v = row[i] * dt[((size_t)b2 * SEQLEN + l) * NHEADS + h];
    Xbf[((((size_t)b2 * NHEADS + h) * NCHUNK + c) * CHUNK + lc) * HEADDIM + p] = (__bf16)v;
  }
  for (int i = threadIdx.x; i < 2 * DSTATE; i += 256) {
    if (i < DSTATE) Bm2[((size_t)b2 * SEQLEN + l) * DSTATE + i] = (__bf16)row[DINNER + i];
    else            Cm2[((size_t)b2 * SEQLEN + l) * DSTATE + (i - DSTATE)] = (__bf16)row[DINNER + i];
  }
}

// =====================================================================
// G[l,s] = C[l,:] . B[s,:]   (shared across heads), WMMA, K=64
// one block per (b2,c); 256 tiles of 16x16 split over 8 waves
// =====================================================================
__global__ __launch_bounds__(256) void k_gmat(
    const __bf16* __restrict__ Cm2, const __bf16* __restrict__ Bm2,
    __bf16* __restrict__ G)
{
  size_t rowbase = (size_t)(blockIdx.x / NCHUNK) * SEQLEN + (blockIdx.x % NCHUNK) * CHUNK;
  size_t gbase = (size_t)blockIdx.x * CHUNK * CHUNK;
  int lane = threadIdx.x & 31, wid = threadIdx.x >> 5;
  int hf = lane >> 4, l15 = lane & 15;
  for (int t = wid; t < 256; t += 8) {
    int lt = t >> 4, st = t & 15;
    f32x8 acc = {};
#pragma unroll
    for (int ks = 0; ks < 2; ++ks) {
      bf16x16 a, b;
      int l = lt * 16 + l15;
      const __bf16* crow = Cm2 + (rowbase + l) * DSTATE;
#pragma unroll
      for (int vv = 0; vv < 8; ++vv) {
        int kk = ks * 32 + ((vv < 4) ? (2 * vv + 8 * hf) : (16 + 2 * (vv - 4) + 8 * hf));
        a[2 * vv] = crow[kk]; a[2 * vv + 1] = crow[kk + 1];
      }
      int s = st * 16 + l15;
      const __bf16* brow = Bm2 + (rowbase + s) * DSTATE;
#pragma unroll
      for (int vv = 0; vv < 8; ++vv) {
        int kk = ks * 32 + ((vv < 4) ? (2 * vv + 8 * hf) : (16 + 2 * (vv - 4) + 8 * hf));
        b[2 * vv] = brow[kk]; b[2 * vv + 1] = brow[kk + 1];
      }
      acc = __builtin_amdgcn_wmma_f32_16x16x32_bf16(false, a, false, b, (short)0, acc, false, false);
    }
#pragma unroll
    for (int r = 0; r < 8; ++r) {
      int l = lt * 16 + r + 8 * hf;
      int s = st * 16 + l15;
      G[gbase + (size_t)l * CHUNK + s] = (__bf16)acc[r];
    }
  }
}

// =====================================================================
// states[p,n] = sum_l decay[l]*X[l,p] * B[l,n]   WMMA, K=256
// one block per (b2,c,h)
// =====================================================================
__global__ __launch_bounds__(256) void k_states(
    const __bf16* __restrict__ Xbf, const __bf16* __restrict__ Bm2,
    const float* __restrict__ Acum, float* __restrict__ states)
{
  int h  = blockIdx.x % NHEADS;
  int c  = (blockIdx.x / NHEADS) % NCHUNK;
  int b2 = blockIdx.x / (NHEADS * NCHUNK);
  __shared__ float dec[CHUNK];
  if (threadIdx.x < CHUNK) {
    float last = Acum[(size_t)blockIdx.x * CHUNK + CHUNK - 1];
    dec[threadIdx.x] = __expf(last - Acum[(size_t)blockIdx.x * CHUNK + threadIdx.x]);
  }
  __syncthreads();
  size_t xb = (((size_t)b2 * NHEADS + h) * NCHUNK + c) * CHUNK * HEADDIM;
  size_t bb = ((size_t)b2 * SEQLEN + c * CHUNK) * DSTATE;
  int lane = threadIdx.x & 31, wid = threadIdx.x >> 5;
  int hf = lane >> 4, l15 = lane & 15;
  for (int t = wid; t < 16; t += 8) {
    int pt = t >> 2, nt = t & 3;
    f32x8 acc = {};
    for (int ks = 0; ks < 8; ++ks) {
      bf16x16 a, b;
      int pr = pt * 16 + l15;
#pragma unroll
      for (int vv = 0; vv < 8; ++vv) {
        int kk = ks * 32 + ((vv < 4) ? (2 * vv + 8 * hf) : (16 + 2 * (vv - 4) + 8 * hf));
        a[2 * vv]     = (__bf16)((float)Xbf[xb + (size_t)kk * HEADDIM + pr] * dec[kk]);
        a[2 * vv + 1] = (__bf16)((float)Xbf[xb + (size_t)(kk + 1) * HEADDIM + pr] * dec[kk + 1]);
      }
      int n = nt * 16 + l15;
#pragma unroll
      for (int vv = 0; vv < 8; ++vv) {
        int kk = ks * 32 + ((vv < 4) ? (2 * vv + 8 * hf) : (16 + 2 * (vv - 4) + 8 * hf));
        b[2 * vv]     = Bm2[bb + (size_t)kk * DSTATE + n];
        b[2 * vv + 1] = Bm2[bb + (size_t)(kk + 1) * DSTATE + n];
      }
      acc = __builtin_amdgcn_wmma_f32_16x16x32_bf16(false, a, false, b, (short)0, acc, false, false);
    }
#pragma unroll
    for (int r = 0; r < 8; ++r) {
      int pr = pt * 16 + r + 8 * hf;
      int n  = nt * 16 + l15;
      states[(size_t)blockIdx.x * 4096 + pr * DSTATE + n] = acc[r];
    }
  }
}

// =====================================================================
// Inter-chunk recurrence: prev_c = R_{c-1}; R_c = exp(Atot_c)*R_{c-1}+states_c
// one block per (b2,h); each thread owns 16 of the 4096 state entries
// =====================================================================
__global__ __launch_bounds__(256) void k_prev(
    const float* __restrict__ states, const float* __restrict__ Acum,
    float* __restrict__ prev)
{
  int b2 = blockIdx.x / NHEADS, h = blockIdx.x % NHEADS;
  float r[16];
#pragma unroll
  for (int j = 0; j < 16; ++j) r[j] = 0.f;
  for (int c = 0; c < NCHUNK; ++c) {
    size_t blk = ((size_t)b2 * NCHUNK + c) * NHEADS + h;
    size_t base = blk * 4096;
    float dec = __expf(Acum[blk * CHUNK + CHUNK - 1]);
    for (int j = 0; j < 16; ++j) {
      int e = threadIdx.x + j * 256;
      prev[base + e] = r[j];
      r[j] = r[j] * dec + states[base + e];
    }
  }
}

// =====================================================================
// Y = (G .* Lmat) @ X + (exp(Acum)*C) @ prev^T    (main SSD kernel, WMMA)
// one block per (b2,c,h); wave w handles l-tiles 2w, 2w+1 (causal K loop)
// =====================================================================
__global__ __launch_bounds__(256) void k_ssd(
    const __bf16* __restrict__ G, const __bf16* __restrict__ Xbf,
    const __bf16* __restrict__ Cm2, const float* __restrict__ prev,
    const float* __restrict__ Acum, float* __restrict__ y4)
{
  int h  = blockIdx.x % NHEADS;
  int c  = (blockIdx.x / NHEADS) % NCHUNK;
  int b2 = blockIdx.x / (NHEADS * NCHUNK);
  __shared__ __bf16 Xs[CHUNK][HEADDIM];     // 32 KB
  __shared__ __bf16 Pv[HEADDIM][HEADDIM];   //  8 KB
  __shared__ float  Ac[CHUNK];              //  1 KB
  size_t xb = (((size_t)b2 * NHEADS + h) * NCHUNK + c) * CHUNK * HEADDIM;
  for (int i = threadIdx.x; i < CHUNK * HEADDIM; i += 256) (&Xs[0][0])[i] = Xbf[xb + i];
  size_t pb = (size_t)blockIdx.x * 4096;
  for (int i = threadIdx.x; i < HEADDIM * HEADDIM; i += 256) (&Pv[0][0])[i] = (__bf16)prev[pb + i];
  if (threadIdx.x < CHUNK) Ac[threadIdx.x] = Acum[(size_t)blockIdx.x * CHUNK + threadIdx.x];
  __syncthreads();
  size_t gb = ((size_t)b2 * NCHUNK + c) * CHUNK * CHUNK;
  size_t crow0 = (size_t)b2 * SEQLEN + c * CHUNK;
  int lane = threadIdx.x & 31, wid = threadIdx.x >> 5;
  int hf = lane >> 4, l15 = lane & 15;
  for (int q = 0; q < 2; ++q) {
    int lt = 2 * wid + q;
    int l = lt * 16 + l15;
    float al = Ac[l];
    float el = __expf(al);                 // <= 1, safe in bf16
    f32x8 acc[4] = {};
    int maxks = (lt * 16 + 15) >> 5;
    for (int ks = 0; ks <= maxks; ++ks) {  // causal K loop over s
      bf16x16 a;
#pragma unroll
      for (int vv = 0; vv < 8; ++vv) {
        int kk = ks * 32 + ((vv < 4) ? (2 * vv + 8 * hf) : (16 + 2 * (vv - 4) + 8 * hf));
        float g0 = (float)G[gb + (size_t)l * CHUNK + kk];
        float g1 = (float)G[gb + (size_t)l * CHUNK + kk + 1];
        float f0 = (kk     <= l) ? __expf(al - Ac[kk])     : 0.f;  // <= 1
        float f1 = (kk + 1 <= l) ? __expf(al - Ac[kk + 1]) : 0.f;
        a[2 * vv] = (__bf16)(g0 * f0); a[2 * vv + 1] = (__bf16)(g1 * f1);
      }
#pragma unroll
      for (int pt = 0; pt < 4; ++pt) {
        bf16x16 b;
        int p = pt * 16 + l15;
#pragma unroll
        for (int vv = 0; vv < 8; ++vv) {
          int s = ks * 32 + ((vv < 4) ? (2 * vv + 8 * hf) : (16 + 2 * (vv - 4) + 8 * hf));
          b[2 * vv] = Xs[s][p]; b[2 * vv + 1] = Xs[s + 1][p];
        }
        acc[pt] = __builtin_amdgcn_wmma_f32_16x16x32_bf16(false, a, false, b, (short)0, acc[pt], false, false);
      }
    }
    // Y_off: A = exp(Acum[l])*C[l,n], B[n,p] = prev[p,n]; K = 64
    const __bf16* crow = Cm2 + (crow0 + l) * DSTATE;
#pragma unroll
    for (int ks = 0; ks < 2; ++ks) {
      bf16x16 a;
#pragma unroll
      for (int vv = 0; vv < 8; ++vv) {
        int kk = ks * 32 + ((vv < 4) ? (2 * vv + 8 * hf) : (16 + 2 * (vv - 4) + 8 * hf));
        a[2 * vv]     = (__bf16)((float)crow[kk]     * el);
        a[2 * vv + 1] = (__bf16)((float)crow[kk + 1] * el);
      }
#pragma unroll
      for (int pt = 0; pt < 4; ++pt) {
        bf16x16 b;
        int p = pt * 16 + l15;
#pragma unroll
        for (int vv = 0; vv < 8; ++vv) {
          int n = ks * 32 + ((vv < 4) ? (2 * vv + 8 * hf) : (16 + 2 * (vv - 4) + 8 * hf));
          b[2 * vv] = Pv[p][n]; b[2 * vv + 1] = Pv[p][n + 1];
        }
        acc[pt] = __builtin_amdgcn_wmma_f32_16x16x32_bf16(false, a, false, b, (short)0, acc[pt], false, false);
      }
    }
#pragma unroll
    for (int pt = 0; pt < 4; ++pt)
#pragma unroll
      for (int r = 0; r < 8; ++r) {
        int ll = c * CHUNK + lt * 16 + r + 8 * hf;
        int p  = pt * 16 + l15;
        y4[((size_t)b2 * SEQLEN + ll) * DINNER + h * HEADDIM + p] = acc[pt][r];
      }
  }
}

// =====================================================================
// roll(+1) both halves, flip the second half, add
// =====================================================================
__global__ __launch_bounds__(256) void k_combine(
    const float* __restrict__ y4, float* __restrict__ y2)
{
  int idx = blockIdx.x * 256 + threadIdx.x;
  if (idx >= BATCHN * SEQLEN * DINNER) return;
  int d = idx % DINNER;
  int l = (idx / DINNER) % SEQLEN;
  int b = idx / (DINNER * SEQLEN);
  float v = (l > 0) ? y4[((size_t)b * SEQLEN + l - 1) * DINNER + d] : 0.f;
  if (l < SEQLEN - 1) v += y4[((size_t)(b + 2) * SEQLEN + (SEQLEN - 2 - l)) * DINNER + d];
  y2[idx] = v;
}

// =====================================================================
// Dterm (24 dots of 1536) + y += x*Dterm_rep + silu(z) gate + RMS norm
// one block per row
// =====================================================================
__global__ __launch_bounds__(256) void k_gate(
    const float* __restrict__ y2, const float* __restrict__ conv,
    const float* __restrict__ zx, const float* __restrict__ fcD,
    const float* __restrict__ Dvec, const float* __restrict__ rms_w,
    float* __restrict__ yfin)
{
  int row = blockIdx.x;
  __shared__ float xsr[DINNER];
  __shared__ float Dterm[NHEADS];
  __shared__ float red[8];
  const float* xr = conv + (size_t)row * CONVDIM;   // xs = cols [0,1536)
  for (int i = threadIdx.x; i < DINNER; i += 256) xsr[i] = xr[i];
  __syncthreads();
  int wid = threadIdx.x >> 5, lane = threadIdx.x & 31;
  for (int h = wid; h < NHEADS; h += 8) {
    float s = 0.f;
    for (int i = lane; i < DINNER; i += 32) s += xsr[i] * fcD[(size_t)h * DINNER + i];
    for (int o = 16; o > 0; o >>= 1) s += __shfl_down(s, o, 32);
    if (lane == 0) Dterm[h] = s + Dvec[h];
  }
  __syncthreads();
  float vals[6]; float ss = 0.f;
#pragma unroll
  for (int j = 0; j < 6; ++j) {
    int i = threadIdx.x + j * 256;
    float v = y2[(size_t)row * DINNER + i] + xsr[i] * Dterm[i >> 6];
    float zz = zx[(size_t)row * DINPROJ + i];
    v *= zz / (1.f + __expf(-zz));
    vals[j] = v; ss += v * v;
  }
  for (int o = 16; o > 0; o >>= 1) ss += __shfl_down(ss, o, 32);
  if (lane == 0) red[wid] = ss;
  __syncthreads();
  if (threadIdx.x == 0) { float t = 0.f; for (int i = 0; i < 8; ++i) t += red[i]; red[0] = t; }
  __syncthreads();
  float scale = rsqrtf(red[0] * (1.f / DINNER) + 1e-5f);
#pragma unroll
  for (int j = 0; j < 6; ++j) {
    int i = threadIdx.x + j * 256;
    yfin[(size_t)row * DINNER + i] = vals[j] * scale * rms_w[i];
  }
}

// =====================================================================
// host launcher
// =====================================================================
extern "C" void kernel_launch(void* const* d_in, const int* in_sizes, int n_in,
                              void* d_out, int out_size, void* d_ws, size_t ws_size,
                              hipStream_t stream) {
  const float* x          = (const float*)d_in[0];
  const float* ln_w       = (const float*)d_in[1];
  const float* ln_b       = (const float*)d_in[2];
  const float* in_proj_w  = (const float*)d_in[3];
  const float* conv_w     = (const float*)d_in[4];
  const float* conv_b     = (const float*)d_in[5];
  const float* dt_bias    = (const float*)d_in[6];
  const float* A_log      = (const float*)d_in[7];
  const float* Dvec       = (const float*)d_in[8];
  const float* fc_D_w     = (const float*)d_in[9];
  const float* rms_w      = (const float*)d_in[10];
  const float* out_proj_w = (const float*)d_in[11];
  float* out = (float*)d_out;

  char* p = (char*)d_ws;
  auto alloc = [&](size_t bytes) -> void* {
    void* q = (void*)p; p += (bytes + 255) & ~(size_t)255; return q;
  };
  float*  u       = (float*) alloc((size_t)ROWS * DMODEL * 4);
  float*  zx      = (float*) alloc((size_t)ROWS * DINPROJ * 4);
  float*  convout = (float*) alloc((size_t)BATCHN * SEQLEN * CONVDIM * 4);
  float*  dt      = (float*) alloc((size_t)NB2 * SEQLEN * NHEADS * 4);
  float*  Acum    = (float*) alloc((size_t)NB2 * NCHUNK * NHEADS * CHUNK * 4);
  __bf16* Xbf     = (__bf16*)alloc((size_t)NB2 * NHEADS * NCHUNK * CHUNK * HEADDIM * 2);
  __bf16* Bm2     = (__bf16*)alloc((size_t)NB2 * SEQLEN * DSTATE * 2);
  __bf16* Cm2     = (__bf16*)alloc((size_t)NB2 * SEQLEN * DSTATE * 2);
  __bf16* G       = (__bf16*)alloc((size_t)NB2 * NCHUNK * CHUNK * CHUNK * 2);
  float*  states  = (float*) alloc((size_t)NB2 * NCHUNK * NHEADS * 4096 * 4);
  float*  prev    = (float*) alloc((size_t)NB2 * NCHUNK * NHEADS * 4096 * 4);
  float*  y4      = (float*) alloc((size_t)NB2 * SEQLEN * DINNER * 4);
  float*  y2      = (float*) alloc((size_t)BATCHN * SEQLEN * DINNER * 4);
  float*  yfin    = (float*) alloc((size_t)BATCHN * SEQLEN * DINNER * 4);

  // 1) LayerNorm
  k_layernorm<<<ROWS, 256, 0, stream>>>(x, ln_w, ln_b, u);
  // 2) in_proj GEMM (WMMA bf16)
  {
    dim3 g((DINPROJ + GBN - 1) / GBN, ROWS / GBM);
    k_gemm_bf16<<<g, 256, 0, stream>>>(u, in_proj_w, nullptr, zx, ROWS, DINPROJ, DMODEL);
  }
  // 3) depthwise causal conv + silu
  k_conv<<<(BATCHN * SEQLEN * CONVDIM + 255) / 256, 256, 0, stream>>>(zx, conv_w, conv_b, convout);
  // 4) dt = softplus
  k_dt<<<(NB2 * SEQLEN * NHEADS + 255) / 256, 256, 0, stream>>>(zx, dt_bias, dt);
  // 5) per-chunk cumsum of dt*A (wave scan)
  k_scan<<<NB2 * NCHUNK * NHEADS / 8, 256, 0, stream>>>(dt, A_log, Acum);
  // 6) pack X (xs*dt) / B / C into bf16 (with batch-flip)
  k_pack<<<NB2 * SEQLEN, 256, 0, stream>>>(convout, dt, Xbf, Bm2, Cm2);
  // 7) G = C B^T per (b2,chunk) (WMMA)
  k_gmat<<<NB2 * NCHUNK, 256, 0, stream>>>(Cm2, Bm2, G);
  // 8) chunk-local states (WMMA)
  k_states<<<NB2 * NCHUNK * NHEADS, 256, 0, stream>>>(Xbf, Bm2, Acum, states);
  // 9) inter-chunk recurrence
  k_prev<<<NB2 * NHEADS, 256, 0, stream>>>(states, Acum, prev);
  // 10) Y = (G.*L) X + exp(Acum) C prev^T (WMMA)
  k_ssd<<<NB2 * NCHUNK * NHEADS, 256, 0, stream>>>(G, Xbf, Cm2, prev, Acum, y4);
  // 11) roll + flip + add
  k_combine<<<(BATCHN * SEQLEN * DINNER + 255) / 256, 256, 0, stream>>>(y4, y2);
  // 12) Dterm + gate + RMS
  k_gate<<<ROWS, 256, 0, stream>>>(y2, convout, zx, fc_D_w, Dvec, rms_w, yfin);
  // 13) out_proj GEMM + residual (WMMA bf16)
  {
    dim3 g(DMODEL / GBN, ROWS / GBM);
    k_gemm_bf16<<<g, 256, 0, stream>>>(yfin, out_proj_w, x, out, ROWS, DMODEL, DINNER);
  }
}
